// StateSpaceModelEventGenerator_84430467104839
// MI455X (gfx1250) — matrix operand order repaired and checked by reading, not compiled, not run on hardware
//
#include <hip/hip_runtime.h>

// ---------------------------------------------------------------------------
// Model constants
// ---------------------------------------------------------------------------
#define BATCH 8
#define CTX 512
#define FR 512
#define IN_DIM 1024
#define ST_DIM 128
#define HOP 512
#define NS 262144            // n_samples
#define HL 16
#define HD 32
#define NCH 8                // scan chunks
#define CH  64               // frames per chunk
#define PI_F 3.14159265358979323846f

typedef __attribute__((ext_vector_type(16))) _Float16 v16h;
typedef __attribute__((ext_vector_type(8)))  _Float16 v8h;
typedef __attribute__((ext_vector_type(8)))  float    v8f;

// ---------------------------------------------------------------------------
// CDNA5 async copy: global -> LDS, 16B per lane, tracked by ASYNCcnt
// ---------------------------------------------------------------------------
__device__ __forceinline__ void async_b128(unsigned lds_off, const void* gptr) {
    asm volatile("global_load_async_to_lds_b128 %0, %1, off"
                 :: "v"(lds_off), "v"((unsigned long long)gptr) : "memory");
}
__device__ __forceinline__ void wait_async0() {
    asm volatile("s_wait_asynccnt 0x0" ::: "memory");
}

// ---------------------------------------------------------------------------
// Hypernet latent:  lat[b][l] = sum_h hv[b][h] * A[h][l]
// ---------------------------------------------------------------------------
__global__ void k_lat(const float* __restrict__ hv, const float* __restrict__ Amat,
                      float* __restrict__ lat) {
    int t = threadIdx.x;
    if (t < BATCH * HL) {
        int b = t >> 4, l = t & 15;
        float v = 0.f;
        #pragma unroll
        for (int h = 0; h < HD; ++h) v += hv[b * HD + h] * Amat[h * HL + l];
        lat[t] = v;
    }
}

// ---------------------------------------------------------------------------
// Hypernet expansion + transpose:  W[b][r][c] = sum_l lat[b][l]*Bm[l][r*C+c]
// stored as Bt[b][c][r] (f16, K-contiguous for WMMA B-fragments).
// Reads Bm ONCE for all 8 batches (streaming roofline: ~113MB total).
// ---------------------------------------------------------------------------
__global__ __launch_bounds__(256)
void k_expand_t(const float* __restrict__ Bm, const float* __restrict__ lat,
                _Float16* __restrict__ out, int R, int C) {
    __shared__ _Float16 tile[BATCH][32][33];
    __shared__ float latS[BATCH][HL];
    int tx = threadIdx.x, ty = threadIdx.y;
    int t = ty * 32 + tx;
    if (t < BATCH * HL) latS[t >> 4][t & 15] = lat[t];
    __syncthreads();
    long D = (long)R * C;
    int c = blockIdx.x * 32 + tx;
    for (int rr = ty; rr < 32; rr += 8) {
        int r = blockIdx.y * 32 + rr;
        long j = (long)r * C + c;
        float f[HL];
        #pragma unroll
        for (int l = 0; l < HL; ++l) f[l] = Bm[(long)l * D + j];
        #pragma unroll
        for (int b = 0; b < BATCH; ++b) {
            float v = 0.f;
            #pragma unroll
            for (int l = 0; l < HL; ++l) v += latS[b][l] * f[l];
            tile[b][rr][tx] = (_Float16)v;
        }
    }
    __syncthreads();
    for (int cc = ty; cc < 32; cc += 8) {
        long o = (long)(blockIdx.x * 32 + cc) * R + blockIdx.y * 32 + tx;
        #pragma unroll
        for (int b = 0; b < BATCH; ++b) out[(long)b * D + o] = tile[b][tx][cc];
    }
}

// Plain expansion for Ws: writes f16 (scan) and f32 (for Ws^64 squarings)
__global__ void k_expand_ws(const float* __restrict__ Bm, const float* __restrict__ lat,
                            _Float16* __restrict__ outH, float* __restrict__ outF, int D) {
    int j = blockIdx.x * 256 + threadIdx.x;
    if (j >= D) return;
    float f[HL];
    #pragma unroll
    for (int l = 0; l < HL; ++l) f[l] = Bm[(long)l * D + j];
    #pragma unroll
    for (int b = 0; b < BATCH; ++b) {
        float v = 0.f;
        #pragma unroll
        for (int l = 0; l < HL; ++l) v += lat[b * HL + l] * f[l];
        outH[(long)b * D + j] = (_Float16)v;
        outF[(long)b * D + j] = v;
    }
}

// 128x128 matrix square (per batch): C = A @ A
__global__ __launch_bounds__(128)
void k_matsq(const float* __restrict__ A, float* __restrict__ C) {
    __shared__ float row[ST_DIM];
    int b = blockIdx.y, r = blockIdx.x, t = threadIdx.x;
    const float* Ab = A + (long)b * ST_DIM * ST_DIM;
    row[t] = Ab[r * ST_DIM + t];
    __syncthreads();
    float acc = 0.f;
    #pragma unroll 8
    for (int k = 0; k < ST_DIM; ++k) acc = fmaf(row[k], Ab[k * ST_DIM + t], acc);
    C[(long)b * ST_DIM * ST_DIM + r * ST_DIM + t] = acc;
}

// ---------------------------------------------------------------------------
// ctrl_h[b][f][c] = control[b][c][f]^2   (f16, LDS tile transpose)
// ---------------------------------------------------------------------------
__global__ void k_ctrl(const float* __restrict__ cs, _Float16* __restrict__ out) {
    __shared__ float tile[32][33];
    int b = blockIdx.z, tx = threadIdx.x, ty = threadIdx.y;
    for (int rr = ty; rr < 32; rr += 8) {
        float v = cs[((long)b * CTX + blockIdx.y * 32 + rr) * FR + blockIdx.x * 32 + tx];
        tile[rr][tx] = v * v;
    }
    __syncthreads();
    for (int rr = ty; rr < 32; rr += 8) {
        out[((long)b * FR + blockIdx.x * 32 + rr) * CTX + blockIdx.y * 32 + tx] =
            (_Float16)tile[tx][rr];
    }
}

// ---------------------------------------------------------------------------
// LDS fragment loaders per CDNA5 16-bit WMMA VGPR striping (CDNA5 ISA 7.12.2)
// Staged tile rows hold 64 halves padded to TPAD=72 (144B = 9*16B: aligned,
// and m*36 mod 64 is a permutation -> conflict-free ds_load_b128).
// ---------------------------------------------------------------------------
#define TPAD 72

__device__ __forceinline__ v16h lds_a_frag(const _Float16* s, int r0, int ksub, int lane) {
    int m  = r0 + (lane & 15);
    int kb = ksub + ((lane >> 4) << 3);     // halves 0..7 -> kb.., 8..15 -> kb+16..
    const _Float16* p = s + m * TPAD + kb;
    v8h lo = *reinterpret_cast<const v8h*>(p);
    v8h hi = *reinterpret_cast<const v8h*>(p + 16);
    v16h f;
    #pragma unroll
    for (int i = 0; i < 8; ++i) { f[i] = lo[i]; f[i + 8] = hi[i]; }
    return f;
}
__device__ __forceinline__ v16h lds_b_frag(const _Float16* s, int n0, int ksub, int lane) {
    int n  = n0 + (lane & 15);
    int kb = ksub + ((lane >> 4) << 3);     // halves 0..7 -> kb.., 8..15 -> kb+16..
    const _Float16* p = s + n * TPAD + kb;
    v8h lo = *reinterpret_cast<const v8h*>(p);
    v8h hi = *reinterpret_cast<const v8h*>(p + 16);
    v16h f;
    #pragma unroll
    for (int i = 0; i < 8; ++i) { f[i] = lo[i]; f[i + 8] = hi[i]; }
    return f;
}

// ---------------------------------------------------------------------------
// Batched GEMM:  C[b](f32 MxN) = A[b](f16 MxK rowmajor) x Bt[b](f16 [N][K]) [+Cin]
// Block tile 64x128, 8 waves (4M x 2N), wave tile 16x64 (4 accumulators).
// K staged 64 per double-buffer step via global_load_async_to_lds_b128:
//   8 WMMAs + 20 ds_loads per barrier pair; copy of stage k+1 overlaps stage k.
// grid (N/128, M/64, BATCH), block 256.
// ---------------------------------------------------------------------------
__global__ __launch_bounds__(256)
void k_gemm_wmma(const _Float16* __restrict__ A, const _Float16* __restrict__ Bt,
                 const float* __restrict__ Cin, float* __restrict__ C,
                 int M, int N, int K, long sA, long sB, long sC) {
    __shared__ _Float16 Abuf[2][64 * TPAD];
    __shared__ _Float16 Bbuf[2][128 * TPAD];
    int b = blockIdx.z;
    const _Float16* Ag = A  + (long)b * sA + (long)(blockIdx.y * 64) * K;
    const _Float16* Bg = Bt + (long)b * sB + (long)(blockIdx.x * 128) * K;
    C += (long)b * sC;
    int t = threadIdx.x;
    int lane = t & 31, wave = t >> 5;
    int wm = wave & 3, wn = wave >> 2;
    int row0 = blockIdx.y * 64 + wm * 16;
    int col0 = blockIdx.x * 128 + wn * 64;

    // per-thread copy slices (16B chunks; 8 chunks per 64-half row)
    const unsigned strideA = 64 * TPAD * 2;     // bytes per A buffer
    const unsigned strideB = 128 * TPAD * 2;    // bytes per B buffer
    const _Float16* gA[2]; unsigned lA[2];
    #pragma unroll
    for (int i = 0; i < 2; ++i) {               // A: 512 chunks / 256 thr
        int cid = t * 2 + i;
        int row = cid >> 3, ch = (cid & 7) << 3;
        gA[i] = Ag + (long)row * K + ch;
        lA[i] = (unsigned)(unsigned long long)(&Abuf[0][0]) + (row * TPAD + ch) * 2;
    }
    const _Float16* gB[4]; unsigned lB[4];
    #pragma unroll
    for (int i = 0; i < 4; ++i) {               // B: 1024 chunks / 256 thr
        int cid = t * 4 + i;
        int row = cid >> 3, ch = (cid & 7) << 3;
        gB[i] = Bg + (long)row * K + ch;
        lB[i] = (unsigned)(unsigned long long)(&Bbuf[0][0]) + (row * TPAD + ch) * 2;
    }

    v8f acc[4];
    int ccol = lane & 15;
    int crw  = (lane >> 4) << 3;
    if (Cin) {
        const float* Ci = Cin + (long)b * sC;
        #pragma unroll
        for (int j = 0; j < 4; ++j)
            #pragma unroll
            for (int r = 0; r < 8; ++r)
                acc[j][r] = Ci[(long)(row0 + crw + r) * N + col0 + j * 16 + ccol];
    } else {
        #pragma unroll
        for (int j = 0; j < 4; ++j)
            #pragma unroll
            for (int r = 0; r < 8; ++r) acc[j][r] = 0.f;
    }

    // prologue: stage k0 = 0 into buffer 0
    #pragma unroll
    for (int i = 0; i < 2; ++i) async_b128(lA[i], gA[i]);
    #pragma unroll
    for (int i = 0; i < 4; ++i) async_b128(lB[i], gB[i]);
    wait_async0();
    __syncthreads();

    int buf = 0;
    for (int k0 = 0; k0 < K; k0 += 64) {
        int nxt = buf ^ 1;
        if (k0 + 64 < K) {
            #pragma unroll
            for (int i = 0; i < 2; ++i) async_b128(lA[i] + nxt * strideA, gA[i] + k0 + 64);
            #pragma unroll
            for (int i = 0; i < 4; ++i) async_b128(lB[i] + nxt * strideB, gB[i] + k0 + 64);
        }
        const _Float16* As = &Abuf[buf][0];
        const _Float16* Bs = &Bbuf[buf][0];
        #pragma unroll
        for (int sub = 0; sub < 64; sub += 32) {
            v16h af = lds_a_frag(As, wm * 16, sub, lane);
            #pragma unroll
            for (int j = 0; j < 4; ++j) {
                v16h bf = lds_b_frag(Bs, wn * 64 + j * 16, sub, lane);
                acc[j] = __builtin_amdgcn_wmma_f32_16x16x32_f16(
                    false, af, false, bf, (short)0, acc[j], false, false);
            }
        }
        wait_async0();        // stage k+1 landed
        __syncthreads();      // all waves done reading 'buf', writes visible
        buf = nxt;
    }
    #pragma unroll
    for (int j = 0; j < 4; ++j)
        #pragma unroll
        for (int r = 0; r < 8; ++r)
            C[(long)(row0 + crw + r) * N + col0 + j * 16 + ccol] = acc[j][r];
}

// f32 -> f16 convert
__global__ void k_f2h(const float* __restrict__ in, _Float16* __restrict__ out, int n) {
    int i = blockIdx.x * 256 + threadIdx.x;
    if (i < n) out[i] = (_Float16)in[i];
}

// ---------------------------------------------------------------------------
// Chunked linear-recurrence scan. Chunk-local pass (inits==null: zero init,
// record end states) and fix-up pass (inits given, write S2h).
// ---------------------------------------------------------------------------
__global__ __launch_bounds__(128)
void k_scan_chunk(const _Float16* __restrict__ WsH, const float* __restrict__ Ui,
                  const float* __restrict__ inits, float* __restrict__ ends,
                  _Float16* __restrict__ S2h) {
    __shared__ _Float16 WsL[ST_DIM * ST_DIM];
    __shared__ float sbuf[2][ST_DIM];
    int c = blockIdx.x, b = blockIdx.y, t = threadIdx.x;
    const _Float16* W = WsH + (long)b * ST_DIM * ST_DIM;
    for (int i = t; i < ST_DIM * ST_DIM; i += 128) WsL[i] = W[i];
    sbuf[0][t] = inits ? inits[((long)b * NCH + c) * ST_DIM + t] : 0.f;
    __syncthreads();
    const float* U = Ui + ((long)b * FR + c * CH) * ST_DIM;
    _Float16* S = S2h ? S2h + ((long)b * FR + c * CH) * ST_DIM : (_Float16*)0;
    int cur = 0;
    float last = 0.f;
    for (int f = 0; f < CH; ++f) {
        const float* sp = sbuf[cur];
        float acc = 0.f;
        #pragma unroll 8
        for (int s = 0; s < ST_DIM; ++s)
            acc = fmaf(sp[s], (float)WsL[s * ST_DIM + t], acc);
        if (S) S[f * ST_DIM + t] = (_Float16)acc;
        last = acc + U[f * ST_DIM + t];
        sbuf[cur ^ 1][t] = last;
        __syncthreads();
        cur ^= 1;
    }
    if (ends) ends[((long)b * NCH + c) * ST_DIM + t] = last;
}

// Sequential 8-chunk combine: s_in(c) = s_in(c-1) @ Ws^64 + end(c-1)
__global__ __launch_bounds__(128)
void k_combine(const float* __restrict__ W64, const float* __restrict__ ends,
               float* __restrict__ inits) {
    __shared__ float s[ST_DIM];
    int b = blockIdx.x, t = threadIdx.x;
    const float* W = W64 + (long)b * ST_DIM * ST_DIM;
    inits[(long)b * NCH * ST_DIM + t] = 0.f;
    s[t] = 0.f;
    __syncthreads();
    for (int c = 1; c < NCH; ++c) {
        float acc = 0.f;
        #pragma unroll 8
        for (int k = 0; k < ST_DIM; ++k) acc = fmaf(s[k], W[k * ST_DIM + t], acc);
        float v = acc + ends[((long)b * NCH + c - 1) * ST_DIM + t];
        inits[((long)b * NCH + c) * ST_DIM + t] = v;
        __syncthreads();
        s[t] = v;
        __syncthreads();
    }
}

// ---------------------------------------------------------------------------
// Softmax over frames
// ---------------------------------------------------------------------------
__global__ __launch_bounds__(512)
void k_softmax(const float* __restrict__ times, float* __restrict__ imp) {
    __shared__ float red[FR];
    int b = blockIdx.x, t = threadIdx.x;
    float v = times[b * FR + t];
    red[t] = v;
    __syncthreads();
    for (int s = 256; s > 0; s >>= 1) { if (t < s) red[t] = fmaxf(red[t], red[t + s]); __syncthreads(); }
    float m = red[0];
    __syncthreads();
    float e = __expf(v - m);
    red[t] = e;
    __syncthreads();
    for (int s = 256; s > 0; s >>= 1) { if (t < s) red[t] += red[t + s]; __syncthreads(); }
    imp[b * FR + t] = e / red[0];
}

// ---------------------------------------------------------------------------
// Overlap-add with periodic Hann window; at most 2 frames touch each sample.
// ---------------------------------------------------------------------------
__global__ void k_ola(const float* __restrict__ Y, float* __restrict__ sig) {
    long i = (long)blockIdx.x * 256 + threadIdx.x;
    if (i >= (long)BATCH * NS) return;
    int b = (int)(i >> 18);
    int n = (int)(i & (NS - 1));
    int r = n & (HOP - 1);
    int f0 = n >> 9;
    const float* Yb = Y + (long)b * FR * IN_DIM;
    const float c = 2.0f * PI_F / (float)IN_DIM;
    float w1 = 0.5f - 0.5f * __cosf(c * (float)r);
    float v = w1 * Yb[(long)f0 * IN_DIM + r];
    if (f0 >= 1) {
        int r2 = r + HOP;
        float w2 = 0.5f - 0.5f * __cosf(c * (float)r2);
        v += w2 * Yb[(long)(f0 - 1) * IN_DIM + r2];
    }
    sig[i] = v;
}

// ---------------------------------------------------------------------------
// Sparse impulse-train convolution (replaces the reference's FFT):
//   out[b][n] = sum_{k=0}^{n>>9} imp[b][k] * sig[b][n-512k]
// ---------------------------------------------------------------------------
__global__ __launch_bounds__(256)
void k_conv(const float* __restrict__ sig, const float* __restrict__ imp,
            float* __restrict__ out) {
    __shared__ float impL[FR];
    int b = blockIdx.y;
    int n = blockIdx.x * 256 + threadIdx.x;
    impL[threadIdx.x]       = imp[b * FR + threadIdx.x];
    impL[256 + threadIdx.x] = imp[b * FR + 256 + threadIdx.x];
    __syncthreads();
    const float* s = sig + (long)b * NS;
    int kmax = n >> 9;
    float acc = 0.f;
    for (int k = 0; k <= kmax; ++k)
        acc = fmaf(impL[k], s[n - (k << 9)], acc);
    out[(long)b * NS + n] = acc;
}

// ---------------------------------------------------------------------------
// Launcher
// ---------------------------------------------------------------------------
extern "C" void kernel_launch(void* const* d_in, const int* in_sizes, int n_in,
                              void* d_out, int out_size, void* d_ws, size_t ws_size,
                              hipStream_t stream) {
    const float* control  = (const float*)d_in[0];
    const float* state_hv = (const float*)d_in[1];
    const float* out_hv   = (const float*)d_in[2];
    const float* in_hv    = (const float*)d_in[3];
    const float* dir_hv   = (const float*)d_in[4];
    const float* proj_hv  = (const float*)d_in[5];
    const float* times    = (const float*)d_in[6];
    const float* A_proj   = (const float*)d_in[7];
    const float* B_proj   = (const float*)d_in[8];
    const float* A_state  = (const float*)d_in[9];
    const float* B_state  = (const float*)d_in[10];
    const float* A_input  = (const float*)d_in[11];
    const float* B_input  = (const float*)d_in[12];
    const float* A_output = (const float*)d_in[13];
    const float* B_output = (const float*)d_in[14];
    const float* A_direct = (const float*)d_in[15];
    const float* B_direct = (const float*)d_in[16];
    float* out = (float*)d_out;

    char* ws = (char*)d_ws;
    size_t off = 0;
    auto carve = [&](size_t bytes) { void* p = ws + off; off += (bytes + 255) & ~(size_t)255; return p; };
    float*    lat   = (float*)   carve(5 * BATCH * HL * 4);
    _Float16* WpT   = (_Float16*)carve((size_t)BATCH * IN_DIM * CTX * 2);
    _Float16* WiT   = (_Float16*)carve((size_t)BATCH * ST_DIM * IN_DIM * 2);
    _Float16* WdT   = (_Float16*)carve((size_t)BATCH * IN_DIM * IN_DIM * 2);
    _Float16* WoT   = (_Float16*)carve((size_t)BATCH * IN_DIM * ST_DIM * 2);
    _Float16* WsH   = (_Float16*)carve((size_t)BATCH * ST_DIM * ST_DIM * 2);
    float*    Ws32  = (float*)   carve((size_t)BATCH * ST_DIM * ST_DIM * 4);
    float*    P0    = (float*)   carve((size_t)BATCH * ST_DIM * ST_DIM * 4);
    float*    P1    = (float*)   carve((size_t)BATCH * ST_DIM * ST_DIM * 4);
    _Float16* ctrlH = (_Float16*)carve((size_t)BATCH * FR * CTX * 2);
    float*    proj  = (float*)   carve((size_t)BATCH * FR * IN_DIM * 4);
    _Float16* projH = (_Float16*)carve((size_t)BATCH * FR * IN_DIM * 2);
    float*    Ui    = (float*)   carve((size_t)BATCH * FR * ST_DIM * 4);
    float*    Yd    = (float*)   carve((size_t)BATCH * FR * IN_DIM * 4);
    _Float16* S2h   = (_Float16*)carve((size_t)BATCH * FR * ST_DIM * 2);
    float*    ends  = (float*)   carve((size_t)BATCH * NCH * ST_DIM * 4);
    float*    inits = (float*)   carve((size_t)BATCH * NCH * ST_DIM * 4);
    float*    imp   = (float*)   carve((size_t)BATCH * FR * 4);
    float*    sig   = (float*)   carve((size_t)BATCH * NS * 4);
    (void)in_sizes; (void)n_in; (void)out_size; (void)ws_size;

    // 1) latents (order: proj, state, input, output, direct)
    k_lat<<<1, 128, 0, stream>>>(proj_hv,  A_proj,   lat + 0 * 128);
    k_lat<<<1, 128, 0, stream>>>(state_hv, A_state,  lat + 1 * 128);
    k_lat<<<1, 128, 0, stream>>>(in_hv,    A_input,  lat + 2 * 128);
    k_lat<<<1, 128, 0, stream>>>(out_hv,   A_output, lat + 3 * 128);
    k_lat<<<1, 128, 0, stream>>>(dir_hv,   A_direct, lat + 4 * 128);

    // 2) hypernet expansions
    dim3 blkT(32, 8);
    k_expand_t<<<dim3(IN_DIM / 32, CTX / 32),    blkT, 0, stream>>>(B_proj,   lat + 0 * 128, WpT, CTX,    IN_DIM);
    k_expand_t<<<dim3(ST_DIM / 32, IN_DIM / 32), blkT, 0, stream>>>(B_input,  lat + 2 * 128, WiT, IN_DIM, ST_DIM);
    k_expand_t<<<dim3(IN_DIM / 32, IN_DIM / 32), blkT, 0, stream>>>(B_direct, lat + 4 * 128, WdT, IN_DIM, IN_DIM);
    k_expand_t<<<dim3(IN_DIM / 32, ST_DIM / 32), blkT, 0, stream>>>(B_output, lat + 3 * 128, WoT, ST_DIM, IN_DIM);
    k_expand_ws<<<(ST_DIM * ST_DIM + 255) / 256, 256, 0, stream>>>(B_state, lat + 1 * 128, WsH, Ws32, ST_DIM * ST_DIM);

    // 3) Ws^64 by repeated squaring
    dim3 gSq(ST_DIM, BATCH);
    k_matsq<<<gSq, 128, 0, stream>>>(Ws32, P0);
    k_matsq<<<gSq, 128, 0, stream>>>(P0, P1);
    k_matsq<<<gSq, 128, 0, stream>>>(P1, P0);
    k_matsq<<<gSq, 128, 0, stream>>>(P0, P1);
    k_matsq<<<gSq, 128, 0, stream>>>(P1, P0);
    k_matsq<<<gSq, 128, 0, stream>>>(P0, P1);   // P1 = Ws^64

    // 4) ctrl^2 transpose -> f16
    k_ctrl<<<dim3(FR / 32, CTX / 32, BATCH), blkT, 0, stream>>>(control, ctrlH);

    // 5) proj = ctrl @ Wp     [512x512 @ 512x1024]
    k_gemm_wmma<<<dim3(IN_DIM / 128, FR / 64, BATCH), 256, 0, stream>>>(
        ctrlH, WpT, nullptr, proj, FR, IN_DIM, CTX,
        (long)FR * CTX, (long)IN_DIM * CTX, (long)FR * IN_DIM);
    k_f2h<<<(BATCH * FR * IN_DIM + 255) / 256, 256, 0, stream>>>(proj, projH, BATCH * FR * IN_DIM);

    // 6) Ui = proj @ Wi ; Yd = proj @ Wd
    k_gemm_wmma<<<dim3(ST_DIM / 128, FR / 64, BATCH), 256, 0, stream>>>(
        projH, WiT, nullptr, Ui, FR, ST_DIM, IN_DIM,
        (long)FR * IN_DIM, (long)ST_DIM * IN_DIM, (long)FR * ST_DIM);
    k_gemm_wmma<<<dim3(IN_DIM / 128, FR / 64, BATCH), 256, 0, stream>>>(
        projH, WdT, nullptr, Yd, FR, IN_DIM, IN_DIM,
        (long)FR * IN_DIM, (long)IN_DIM * IN_DIM, (long)FR * IN_DIM);

    // 7) chunk-parallel linear scan: local pass -> combine -> fix-up pass
    dim3 gSc(NCH, BATCH);
    k_scan_chunk<<<gSc, 128, 0, stream>>>(WsH, Ui, nullptr, ends, nullptr);
    k_combine<<<BATCH, 128, 0, stream>>>(P1, ends, inits);
    k_scan_chunk<<<gSc, 128, 0, stream>>>(WsH, Ui, inits, nullptr, S2h);

    // 8) Y = S2 @ Wo + Yd   (in place into Yd)
    k_gemm_wmma<<<dim3(IN_DIM / 128, FR / 64, BATCH), 256, 0, stream>>>(
        S2h, WoT, Yd, Yd, FR, IN_DIM, ST_DIM,
        (long)FR * ST_DIM, (long)IN_DIM * ST_DIM, (long)FR * IN_DIM);

    // 9) impulse softmax, overlap-add, sparse convolution
    k_softmax<<<BATCH, FR, 0, stream>>>(times, imp);
    k_ola<<<(BATCH * NS + 255) / 256, 256, 0, stream>>>(Yd, sig);
    k_conv<<<dim3(NS / 256, BATCH), 256, 0, stream>>>(sig, imp, out);
}
